// LSTM_41205916238290
// MI455X (gfx1250) — matrix operand order repaired and checked by reading, not compiled, not run on hardware
//
#include <hip/hip_runtime.h>

// ---------------------------------------------------------------------------
// 2-layer LSTM (S=4096, B=64, I=14, H=256) + ReLU + FC(H->1) for MI455X.
//
// Persistent 8-workgroup kernel: WGs 0..3 run layer 0 at step u, WGs 4..7 run
// layer 1 at step u-1 (pipelined). Per step each layer does one bf16 WMMA GEMM
//   gates[64,1024] = [h | input] @ [W_hh^T ; W_ih^T]   (f32 accumulation)
// with all weight B-fragments register-resident (~256 VGPRs/wave, using the
// gfx1250 1024-VGPR/wave budget via VGPR-MSB addressing). c-state stays fp32
// in VGPRs. h circulates through small double-buffered bf16 rings in scratch;
// workgroups sync per step with a global atomic counter barrier.
// Gate columns are interleaved p = 4*j + gate, so i/f/g/o of one hidden unit
// sit in one aligned lane-quad -> combined with v_mov_b32_dpp quad_perm
// broadcasts (no LDS round trip). Hot-loop global accesses use explicit
// address_space(1) pointers so they lower to global_load/global_store (pure
// LOADcnt/STOREcnt) instead of flat (which also ties up DScnt).
// ---------------------------------------------------------------------------

#define S_LEN 4096
#define BATCH 64
#define IDIM  14
#define HDIM  256

typedef __attribute__((ext_vector_type(16))) __bf16 v16bf;
typedef __attribute__((ext_vector_type(8)))  __bf16 v8bf;
typedef __attribute__((ext_vector_type(8)))  float  v8f;

// Global-address-space pointer types (force global_load / global_store).
typedef __attribute__((address_space(1))) const __bf16 gcbf;
typedef __attribute__((address_space(1)))       __bf16 gbf;
typedef __attribute__((address_space(1))) const v8bf   gcv8bf;
typedef __attribute__((address_space(1)))       v8bf   gv8bf;
typedef __attribute__((address_space(1))) const float  gcfloat;
typedef __attribute__((address_space(1)))       float  gfloat;

__device__ __forceinline__ float rcp_fast(float x) { return __builtin_amdgcn_rcpf(x); }
__device__ __forceinline__ float sigm(float x) { return rcp_fast(1.0f + __expf(-x)); }
__device__ __forceinline__ float tanh_fast(float x) {
  // 1 - 2/(e^{2x}+1): exp->inf gives +1, exp->0 gives -1.
  return 1.0f - 2.0f * rcp_fast(__expf(2.0f * x) + 1.0f);
}
// Broadcast the value held by lane (quad_base + QL) to all 4 lanes of a quad.
template<int CTRL>
__device__ __forceinline__ float quad_bcast(float v) {
  return __int_as_float(
      __builtin_amdgcn_update_dpp(0, __float_as_int(v), CTRL, 0xF, 0xF, true));
}

// LAYER 0: NKT = 9  (K = 256 (h) + 32 (x padded))
// LAYER 1: NKT = 16 (K = 256 (h1) + 256 (h0 stream))
template<int LAYER, int NKT>
__device__ void lstm_layer_run(
    const float* __restrict__ x_f32,     // layer0 only: x [S,B,I]
    const float* __restrict__ c0,        // [2,B,H]
    const float* __restrict__ W_hh,      // [1024,256]
    const float* __restrict__ W_ih,      // [1024, I or 256]
    const float* __restrict__ b_ih,
    const float* __restrict__ b_hh,
    const float* __restrict__ wfc,       // layer1 only [256]
    unsigned short* ring_in_us,          // layer1: h0 ring (input stream)
    unsigned short* ring_h_us,           // this layer's own h ring [2][B][H]
    float* __restrict__ partials,        // layer1 only [S][4][64]
    unsigned* counter,
    int jblk)                            // which 64-hidden-unit slice (0..3)
{
  __shared__ __bf16 hnext[BATCH][64];    // this WG's new-h slice
  __shared__ float  wfc_s[64];

  gbf*  ring_h  = (gbf*)ring_h_us;
  gcbf* ring_in = (gcbf*)ring_in_us;
  gcfloat* xg   = (gcfloat*)x_f32;

  const int tid   = threadIdx.x;
  const int w     = tid >> 5;        // wave 0..7
  const int L     = tid & 31;        // lane
  const int half  = L >> 4;          // lane half (K-half for A/B frags)
  const int n16   = L & 15;          // N within tile / M within A tile
  const int gate  = n16 & 3;         // column interleave: p = 4*j + gate
  const int qq    = n16 >> 2;
  const int jbase = jblk * 64;

  // ---- one-time setup: pack B fragments (weights) into VGPRs --------------
  int   jl_tt[2];
  float biasv[2];
  v16bf bfrag[NKT][2];

  #pragma unroll
  for (int tt = 0; tt < 2; ++tt) {
    const int tile = 2 * w + tt;           // N-tile within WG's 256 columns
    const int jl   = 4 * tile + qq;        // local hidden unit 0..63
    jl_tt[tt] = jl;
    const int row = gate * HDIM + jbase + jl;   // gate row in [0,1024)
    biasv[tt] = ((gcfloat*)b_ih)[row] + ((gcfloat*)b_hh)[row];
    #pragma unroll
    for (int kt = 0; kt < NKT; ++kt) {
      v16bf bf;
      #pragma unroll
      for (int e = 0; e < 16; ++e) {
        // B 32x16 bf16 layout: elems 0..7 = K 8*half.., 8..15 = K 16+8*half..
        const int K = kt * 32 + ((e < 8) ? (8 * half + e) : (16 + 8 * half + (e - 8)));
        float v;
        if (LAYER == 0) {
          if (K < HDIM) v = ((gcfloat*)W_hh)[row * HDIM + K];
          else { const int ki = K - HDIM;
                 v = (ki < IDIM) ? ((gcfloat*)W_ih)[row * IDIM + ki] : 0.0f; }
        } else {
          v = (K < HDIM) ? ((gcfloat*)W_hh)[row * HDIM + K]
                         : ((gcfloat*)W_ih)[row * HDIM + (K - HDIM)];
        }
        bf[e] = (__bf16)v;
      }
      bfrag[kt][tt] = bf;
    }
  }

  // ---- fp32 c-state in VGPRs (redundant across the 4 gate lanes) ----------
  float creg[4][2][8];
  #pragma unroll
  for (int mt = 0; mt < 4; ++mt)
    #pragma unroll
    for (int tt = 0; tt < 2; ++tt)
      #pragma unroll
      for (int r = 0; r < 8; ++r) {
        const int m = 16 * mt + 8 * half + r;        // D-layout row
        creg[mt][tt][r] = ((gcfloat*)c0)[(LAYER * BATCH + m) * HDIM + jbase + jl_tt[tt]];
      }

  if (LAYER == 1 && tid < 64) wfc_s[tid] = ((gcfloat*)wfc)[jbase + tid];
  __syncthreads();

  // ---- persistent time loop (layer1 lags one step) ------------------------
  for (int u = 0; u <= S_LEN; ++u) {
    const bool active = (LAYER == 0) ? (u < S_LEN) : (u >= 1);
    if (active) {
      const int s = (LAYER == 0) ? u : (u - 1);
      gcbf* hprev = (gcbf*)ring_h + ((s + 1) & 1) * (BATCH * HDIM);
      gcbf* xprev = (LAYER == 1) ? (ring_in + (s & 1) * (BATCH * HDIM)) : (gcbf*)0;

      v8f acc[4][2];
      #pragma unroll
      for (int mt = 0; mt < 4; ++mt)
        #pragma unroll
        for (int tt = 0; tt < 2; ++tt) {
          v8f z = {0.f, 0.f, 0.f, 0.f, 0.f, 0.f, 0.f, 0.f};
          acc[mt][tt] = z;
        }

      #pragma unroll
      for (int kt = 0; kt < NKT; ++kt) {
        v16bf afr[4];
        #pragma unroll
        for (int mt = 0; mt < 4; ++mt) {
          const int m = 16 * mt + n16;   // A layout: lane%16 = M row
          if (LAYER == 0 && kt == NKT - 1) {
            // x tile: K-256 columns 0..13 = x[s], rest zero.
            v16bf a;
            #pragma unroll
            for (int e = 0; e < 16; ++e) {
              const int ki = (e < 8) ? (8 * half + e) : (16 + 8 * half + (e - 8));
              const float v = (ki < IDIM)
                  ? xg[((size_t)s * BATCH + m) * IDIM + ki] : 0.0f;
              a[e] = (__bf16)v;
            }
            afr[mt] = a;
          } else {
            // A bf16 16x32 layout: elems 0..7 = K k0..k0+7, 8..15 = k0+16..
            gcbf* src = (LAYER == 1 && kt >= 8)
                ? (xprev + m * HDIM + (kt - 8) * 32)
                : (hprev + m * HDIM + kt * 32);
            v8bf lo = *(gcv8bf*)(src + 8 * half);
            v8bf hi = *(gcv8bf*)(src + 16 + 8 * half);
            afr[mt] = __builtin_shufflevector(lo, hi,
                        0,1,2,3,4,5,6,7,8,9,10,11,12,13,14,15);
          }
        }
        #pragma unroll
        for (int tt = 0; tt < 2; ++tt)
          #pragma unroll
          for (int mt = 0; mt < 4; ++mt)
            acc[mt][tt] = __builtin_amdgcn_wmma_f32_16x16x32_bf16(
                false, afr[mt], false, bfrag[kt][tt],
                (short)0, acc[mt][tt], false, false);
      }

      // ---- gate nonlinearities + state update -----------------------------
      // i,f,g,o of one hidden unit live in one aligned lane-quad: combine
      // with DPP quad_perm broadcasts (pure VALU, no LDS).
      #pragma unroll
      for (int mt = 0; mt < 4; ++mt)
        #pragma unroll
        for (int tt = 0; tt < 2; ++tt)
          #pragma unroll
          for (int r = 0; r < 8; ++r) {
            const float v   = acc[mt][tt][r] + biasv[tt];
            const float act = (gate == 2) ? tanh_fast(v) : sigm(v);
            const float iv = quad_bcast<0x00>(act);  // quad_perm [0,0,0,0]
            const float fv = quad_bcast<0x55>(act);  // quad_perm [1,1,1,1]
            const float gv = quad_bcast<0xAA>(act);  // quad_perm [2,2,2,2]
            const float ov = quad_bcast<0xFF>(act);  // quad_perm [3,3,3,3]
            const float cn = fv * creg[mt][tt][r] + iv * gv;
            creg[mt][tt][r] = cn;
            const float hv = ov * tanh_fast(cn);
            if (gate == 0) {
              const int m = 16 * mt + 8 * half + r;
              hnext[m][jl_tt[tt]] = (__bf16)hv;
            }
          }
      __syncthreads();

      // ---- publish h slice (coalesced) + FC partial for layer1 ------------
      {
        gbf* hout = ring_h + (s & 1) * (BATCH * HDIM);
        const int b = tid >> 2, seg = tid & 3;
        gv8bf* dst = (gv8bf*)(hout + b * HDIM + jbase + seg * 16);
        const v8bf* sp = (const v8bf*)(&hnext[b][seg * 16]);
        dst[0] = sp[0];
        dst[1] = sp[1];
        if (LAYER == 1) {
          float sum = 0.0f;
          #pragma unroll
          for (int j = 0; j < 16; ++j) {
            const float hv = (float)hnext[b][seg * 16 + j];
            sum += fmaxf(hv, 0.0f) * wfc_s[seg * 16 + j];
          }
          // quad reduction via DPP xor-perms [1,0,3,2] and [2,3,0,1]
          sum += __int_as_float(__builtin_amdgcn_update_dpp(
                     0, __float_as_int(sum), 0xB1, 0xF, 0xF, true));
          sum += __int_as_float(__builtin_amdgcn_update_dpp(
                     0, __float_as_int(sum), 0x4E, 0xF, 0xF, true));
          if (seg == 0)
            ((gfloat*)partials)[((size_t)s * 4 + jblk) * BATCH + b] = sum;
        }
      }
    }

    // ---- device-wide step barrier across all 8 workgroups -----------------
    if (u < S_LEN) {
      __threadfence();            // release: h-slice stores visible at L2
      __syncthreads();
      if (tid == 0) {
        __hip_atomic_fetch_add(counter, 1u, __ATOMIC_RELEASE,
                               __HIP_MEMORY_SCOPE_AGENT);
        const unsigned target = 8u * (unsigned)(u + 1);
        while (__hip_atomic_load(counter, __ATOMIC_ACQUIRE,
                                 __HIP_MEMORY_SCOPE_AGENT) < target)
          __builtin_amdgcn_s_sleep(1);
      }
      __syncthreads();
      __threadfence();            // acquire: invalidate near caches
    }
  }
}

__global__ __launch_bounds__(256, 1)
void lstm_persist_kernel(
    const float* x, const float* c0,
    const float* W_ih0, const float* W_hh0, const float* b_ih0, const float* b_hh0,
    const float* W_ih1, const float* W_hh1, const float* b_ih1, const float* b_hh1,
    const float* W_fc,
    unsigned short* ring0, unsigned short* ring1,
    float* partials, unsigned* counter)
{
  const int blk = blockIdx.x;
  if (blk < 4)
    lstm_layer_run<0, 9>(x, c0, W_hh0, W_ih0, b_ih0, b_hh0, nullptr,
                         nullptr, ring0, nullptr, counter, blk);
  else
    lstm_layer_run<1, 16>(nullptr, c0, W_hh1, W_ih1, b_ih1, b_hh1, W_fc,
                          ring0, ring1, partials, counter, blk - 4);
}

__global__ void lstm_init_kernel(const float* __restrict__ h0,
                                 unsigned* counter,
                                 unsigned short* ring0, unsigned short* ring1)
{
  const int tid = blockIdx.x * blockDim.x + threadIdx.x;
  if (tid == 0) *counter = 0u;
  const int n = BATCH * HDIM;
  if (tid < n) {   // step 0 reads ring parity 1
    ((gbf*)ring0)[n + tid] = (__bf16)((gcfloat*)h0)[tid];
    ((gbf*)ring1)[n + tid] = (__bf16)((gcfloat*)h0)[n + tid];
  }
}

__global__ void lstm_fc_kernel(const float* __restrict__ partials,
                               const float* __restrict__ b_fc,
                               float* __restrict__ out)
{
  const int idx = blockIdx.x * blockDim.x + threadIdx.x;
  if (idx < S_LEN * BATCH) {
    const int s = idx >> 6, b = idx & 63;
    gcfloat* p = (gcfloat*)partials + (size_t)s * 256;
    ((gfloat*)out)[idx] = p[b] + p[64 + b] + p[128 + b] + p[192 + b]
                          + ((gcfloat*)b_fc)[0];
  }
}

extern "C" void kernel_launch(void* const* d_in, const int* in_sizes, int n_in,
                              void* d_out, int out_size, void* d_ws, size_t ws_size,
                              hipStream_t stream)
{
  (void)in_sizes; (void)n_in; (void)out_size; (void)ws_size;
  const float* x     = (const float*)d_in[0];
  const float* h0    = (const float*)d_in[1];
  const float* c0    = (const float*)d_in[2];
  const float* W_ih0 = (const float*)d_in[3];
  const float* W_hh0 = (const float*)d_in[4];
  const float* b_ih0 = (const float*)d_in[5];
  const float* b_hh0 = (const float*)d_in[6];
  const float* W_ih1 = (const float*)d_in[7];
  const float* W_hh1 = (const float*)d_in[8];
  const float* b_ih1 = (const float*)d_in[9];
  const float* b_hh1 = (const float*)d_in[10];
  const float* W_fc  = (const float*)d_in[11];
  const float* b_fc  = (const float*)d_in[12];

  char* ws = (char*)d_ws;
  unsigned*       counter = (unsigned*)ws;
  unsigned short* ring0   = (unsigned short*)(ws + 256);
  unsigned short* ring1   = (unsigned short*)(ws + 256 + 2 * BATCH * HDIM * 2);
  float*          partials= (float*)(ws + 256 + 4 * BATCH * HDIM * 2);

  hipLaunchKernelGGL(lstm_init_kernel, dim3((BATCH * HDIM + 255) / 256), dim3(256),
                     0, stream, h0, counter, ring0, ring1);
  hipLaunchKernelGGL(lstm_persist_kernel, dim3(8), dim3(256), 0, stream,
                     x, c0, W_ih0, W_hh0, b_ih0, b_hh0,
                     W_ih1, W_hh1, b_ih1, b_hh1, W_fc,
                     ring0, ring1, partials, counter);
  hipLaunchKernelGGL(lstm_fc_kernel, dim3((S_LEN * BATCH + 255) / 256), dim3(256),
                     0, stream, partials, b_fc, (float*)d_out);
}
